// MultiSemanticHyperConvLayer_23742579212950
// MI455X (gfx1250) — matrix operand order repaired and checked by reading, not compiled, not run on hardware
//
#include <hip/hip_runtime.h>

// ---------------------------------------------------------------------------
// MultiSemanticHyperConvLayer for MI455X (gfx1250, wave32).
//
// Pipeline:
//   K0: transpose+convert fusion_W [896,128] f32 -> Wt [128,896] bf16 (L2-hot B)
//   K1: fused 3-way SpMM over HG_up: one index/val read, three gathers
//       (tables total 153.6MB -> L2-resident), f32 atomic scatter-add.
//   K2: per 32-user tile: build msg [32,896] in LDS as bf16 (7 feature groups),
//       GEMM with v_wmma_f32_16x16x32_bf16 (K=896 = 28 steps), fused
//       bias + (x + u + x*u) gating epilogue -> hg_users [50k,128] f32.
//   K3: SpMM over HG_pu from L2-resident hg_users -> d_out atomics.
// ---------------------------------------------------------------------------

#define EMB 128
#define K7  (7 * EMB)        // 896
#define GEMM_M 32            // users per block in the fusion GEMM

typedef __bf16 bf16_t;
typedef __attribute__((ext_vector_type(16))) __bf16 v16bf;
typedef __attribute__((ext_vector_type(8)))  __bf16 v8bf;
typedef __attribute__((ext_vector_type(8)))  float  v8f;

// ---------------------------------------------------------------------------
// K0: Wt[n][k] = (bf16) W[k][n]   (N-major so B fragments are contiguous 32B)
// ---------------------------------------------------------------------------
__global__ __launch_bounds__(256) void convert_w_kernel(
    const float* __restrict__ W, bf16_t* __restrict__ Wt) {
  int idx = blockIdx.x * 256 + threadIdx.x;      // n-major over 128*896
  if (idx >= EMB * K7) return;
  int n = idx / K7;
  int k = idx - n * K7;
  Wt[idx] = (bf16_t)W[k * EMB + n];
}

// ---------------------------------------------------------------------------
// K1: fused 3-semantic SpMM for HG_up.  One wave per nonzero; lane l owns
// dims [4l,4l+4).  Single read of (row,col,val), three float4 gathers,
// 12 global f32 atomic adds (global_atomic_add_f32 via unsafeAtomicAdd).
// ---------------------------------------------------------------------------
__global__ __launch_bounds__(256) void spmm3_up_kernel(
    const int* __restrict__ rows, const int* __restrict__ cols,
    const float* __restrict__ vals,
    const float* __restrict__ geo, const float* __restrict__ seq,
    const float* __restrict__ poi,
    float* __restrict__ mg, float* __restrict__ ms, float* __restrict__ mp,
    int nnz) {
  int e    = (blockIdx.x * 256 + threadIdx.x) >> 5;   // nonzero index
  int lane = threadIdx.x & 31;
  if (e >= nnz) return;
  int   r = rows[e];
  int   c = cols[e];
  float v = vals[e];
  size_t src = (size_t)c * EMB + lane * 4;
  size_t dst = (size_t)r * EMB + lane * 4;
  float4 g = *(const float4*)(geo + src);
  float4 s = *(const float4*)(seq + src);
  float4 p = *(const float4*)(poi + src);
  float* og = mg + dst; float* os = ms + dst; float* op = mp + dst;
  unsafeAtomicAdd(og + 0, v * g.x); unsafeAtomicAdd(og + 1, v * g.y);
  unsafeAtomicAdd(og + 2, v * g.z); unsafeAtomicAdd(og + 3, v * g.w);
  unsafeAtomicAdd(os + 0, v * s.x); unsafeAtomicAdd(os + 1, v * s.y);
  unsafeAtomicAdd(os + 2, v * s.z); unsafeAtomicAdd(os + 3, v * s.w);
  unsafeAtomicAdd(op + 0, v * p.x); unsafeAtomicAdd(op + 1, v * p.y);
  unsafeAtomicAdd(op + 2, v * p.z); unsafeAtomicAdd(op + 3, v * p.w);
}

// ---------------------------------------------------------------------------
// K2: fusion GEMM + gating.  Block = 256 threads = 8 waves.
// Wave w owns output columns [16w,16w+16); block owns 32 user rows.
// A tile (32 x 896 bf16) staged in LDS with the 7 feature groups built on
// the fly.  Per k-step: B frag = one 32B global load (L2-hot Wt),
// A frags = two 16B LDS loads each; two v_wmma_f32_16x16x32_bf16 per step.
// ---------------------------------------------------------------------------
__global__ __launch_bounds__(256) void fusion_gemm_kernel(
    const float* __restrict__ mg, const float* __restrict__ ms,
    const float* __restrict__ mp,
    const bf16_t* __restrict__ Wt, const float* __restrict__ bias,
    const float* __restrict__ users, float* __restrict__ hg, int n_users) {
  __shared__ __align__(16) bf16_t Atile[GEMM_M][K7];

  const int tid   = threadIdx.x;
  const int ubase = blockIdx.x * GEMM_M;

  // ---- stage A tile: 32 users x 128 dims -> 7 feature groups (bf16) ----
  for (int idx = tid; idx < GEMM_M * EMB; idx += 256) {
    int r = idx >> 7;          // user row in tile
    int d = idx & (EMB - 1);   // embedding dim
    int u = ubase + r;
    float g = 0.f, s = 0.f, p = 0.f;
    if (u < n_users) {
      size_t o = (size_t)u * EMB + d;
      g = mg[o]; s = ms[o]; p = mp[o];
    }
    Atile[r][0 * EMB + d] = (bf16_t)g;
    Atile[r][1 * EMB + d] = (bf16_t)s;
    Atile[r][2 * EMB + d] = (bf16_t)p;
    Atile[r][3 * EMB + d] = (bf16_t)(g * s);
    Atile[r][4 * EMB + d] = (bf16_t)(g * p);
    Atile[r][5 * EMB + d] = (bf16_t)(s * p);
    Atile[r][6 * EMB + d] = (bf16_t)(g * s * p);
  }
  __syncthreads();

  const int wave  = tid >> 5;
  const int lane  = tid & 31;
  const int nloc  = lane & 15;   // N within 16-wide tile (and A row M=lane%16)
  const int khalf = lane >> 4;   // K half select (this code's contiguous k = 16*khalf + s)
  const int ncol  = wave * 16 + nloc;

  v8f acc0 = {};                 // user rows 0..15 of the tile
  v8f acc1 = {};                 // user rows 16..31

  const bf16_t* wrow = Wt + (size_t)ncol * K7;

  for (int kb = 0; kb < K7; kb += 32) {
    // B frag: lane holds B[K][N] for K = kb + 16*khalf + (0..15), N = ncol
    v16bf bfrag = *(const v16bf*)(wrow + kb + 16 * khalf);

    // A frag rows m and m+16: K = kb + 16*khalf + (0..15)
    const bf16_t* a0p = &Atile[nloc][kb + 16 * khalf];
    v8bf a0lo = *(const v8bf*)(a0p);
    v8bf a0hi = *(const v8bf*)(a0p + 8);
    v16bf afrag0 = __builtin_shufflevector(a0lo, a0hi,
        0, 1, 2, 3, 4, 5, 6, 7, 8, 9, 10, 11, 12, 13, 14, 15);

    const bf16_t* a1p = &Atile[nloc + 16][kb + 16 * khalf];
    v8bf a1lo = *(const v8bf*)(a1p);
    v8bf a1hi = *(const v8bf*)(a1p + 8);
    v16bf afrag1 = __builtin_shufflevector(a1lo, a1hi,
        0, 1, 2, 3, 4, 5, 6, 7, 8, 9, 10, 11, 12, 13, 14, 15);

    acc0 = __builtin_amdgcn_wmma_f32_16x16x32_bf16(
        false, afrag0, false, bfrag, (short)0, acc0, false, false);
    acc1 = __builtin_amdgcn_wmma_f32_16x16x32_bf16(
        false, afrag1, false, bfrag, (short)0, acc1, false, false);
  }

  // ---- fused epilogue: msg_emb = acc + b;  hg = m + u + m*u ----
  float bn = bias[ncol];
#pragma unroll
  for (int v = 0; v < 8; ++v) {
    int r0 = v + 8 * khalf;                 // C layout: M = v + 8*(lane/16)
    int u0 = ubase + r0;
    if (u0 < n_users) {
      float me = acc0[v] + bn;
      float ue = users[(size_t)u0 * EMB + ncol];
      hg[(size_t)u0 * EMB + ncol] = me + ue + me * ue;
    }
    int u1 = u0 + 16;
    if (u1 < n_users) {
      float me = acc1[v] + bn;
      float ue = users[(size_t)u1 * EMB + ncol];
      hg[(size_t)u1 * EMB + ncol] = me + ue + me * ue;
    }
  }
}

// ---------------------------------------------------------------------------
// K3: final SpMM over HG_pu: out[r] += v * hg[c]  (hg is L2-resident, 25.6MB)
// ---------------------------------------------------------------------------
__global__ __launch_bounds__(256) void spmm_pu_kernel(
    const int* __restrict__ rows, const int* __restrict__ cols,
    const float* __restrict__ vals,
    const float* __restrict__ hg, float* __restrict__ out, int nnz) {
  int e    = (blockIdx.x * 256 + threadIdx.x) >> 5;
  int lane = threadIdx.x & 31;
  if (e >= nnz) return;
  int   r = rows[e];
  int   c = cols[e];
  float v = vals[e];
  float4 h = *(const float4*)(hg + (size_t)c * EMB + lane * 4);
  float* o = out + (size_t)r * EMB + lane * 4;
  unsafeAtomicAdd(o + 0, v * h.x);
  unsafeAtomicAdd(o + 1, v * h.y);
  unsafeAtomicAdd(o + 2, v * h.z);
  unsafeAtomicAdd(o + 3, v * h.w);
}

// ---------------------------------------------------------------------------
extern "C" void kernel_launch(void* const* d_in, const int* in_sizes, int n_in,
                              void* d_out, int out_size, void* d_ws, size_t ws_size,
                              hipStream_t stream) {
  const float* col_pois = (const float*)d_in[0];
  const float* geo_pois = (const float*)d_in[1];
  const float* seq_pois = (const float*)d_in[2];
  const float* users    = (const float*)d_in[3];
  const int*   up_rows  = (const int*)d_in[4];
  const int*   up_cols  = (const int*)d_in[5];
  const float* up_vals  = (const float*)d_in[6];
  const int*   pu_rows  = (const int*)d_in[7];
  const int*   pu_cols  = (const int*)d_in[8];
  const float* pu_vals  = (const float*)d_in[9];
  const float* fusion_W = (const float*)d_in[10];
  const float* fusion_b = (const float*)d_in[11];

  const int n_users = in_sizes[3] / EMB;   // 50000
  const int nnz_up  = in_sizes[4];         // 2.5M
  const int nnz_pu  = in_sizes[7];         // 2.5M

  // ---- workspace layout (all offsets stay 256B aligned) ----
  const size_t msgElems = (size_t)n_users * EMB;      // 6.4M floats
  float*  mg = (float*)d_ws;
  float*  ms = mg + msgElems;
  float*  mp = ms + msgElems;
  float*  hg = mp + msgElems;
  bf16_t* Wt = (bf16_t*)(hg + msgElems);              // 128*896 bf16

  // zero the three message accumulators and the output (atomic targets)
  hipMemsetAsync(mg, 0, 3 * msgElems * sizeof(float), stream);
  hipMemsetAsync(d_out, 0, (size_t)out_size * sizeof(float), stream);

  // K0: W -> Wt (bf16, N-major)
  convert_w_kernel<<<(EMB * K7 + 255) / 256, 256, 0, stream>>>(fusion_W, Wt);

  // K1: fused 3-way SpMM over HG_up (8 nonzeros per 256-thread block)
  spmm3_up_kernel<<<(nnz_up + 7) / 8, 256, 0, stream>>>(
      up_rows, up_cols, up_vals, geo_pois, seq_pois, col_pois,
      mg, ms, mp, nnz_up);

  // K2: WMMA fusion GEMM + gating
  fusion_gemm_kernel<<<(n_users + GEMM_M - 1) / GEMM_M, 256, 0, stream>>>(
      mg, ms, mp, Wt, fusion_b, users, hg, n_users);

  // K3: final SpMM into d_out
  spmm_pu_kernel<<<(nnz_pu + 7) / 8, 256, 0, stream>>>(
      pu_rows, pu_cols, pu_vals, hg, (float*)d_out, nnz_pu);
}
